// CrossAttention_12343736009291
// MI455X (gfx1250) — matrix operand order repaired and checked by reading, not compiled, not run on hardware
//
#include <hip/hip_runtime.h>

// ---------------------------------------------------------------------------
// MI455X (gfx1250) implementation of the two-stream cross-attention block.
// All heavy GEMMs + attention use v_wmma_f32_16x16x32_bf16 (wave32 WMMA,
// fp32 accumulation). bf16 tile staging uses the CDNA5 async DMA path
// (global_load_async_to_lds_b128 + s_wait_asynccnt); fragment reads are
// per-lane contiguous so they lower to ds_load_b128.
// ---------------------------------------------------------------------------

typedef __attribute__((ext_vector_type(16))) __bf16 v16bf;
typedef __attribute__((ext_vector_type(8)))  float  v8f;

#define DEVINL __device__ __forceinline__

DEVINL v8f wmma_bf16(v16bf a, v16bf b, v8f c) {
  // 8 args: (neg_a, A, neg_b, B, c_mod, C, reuse_a, reuse_b)
  return __builtin_amdgcn_wmma_f32_16x16x32_bf16(false, a, false, b, (short)0, c,
                                                 false, false);
}

// Async global->LDS 16B copy per lane (ASYNCcnt-tracked DMA, no VGPR staging).
DEVINL void async_load_b128(const void* lds_dst, const void* gsrc) {
  unsigned           loff  = (unsigned)(uintptr_t)lds_dst;  // LDS byte offset
  unsigned long long gaddr = (unsigned long long)(uintptr_t)gsrc;
  asm volatile("global_load_async_to_lds_b128 %0, %1, off"
               :
               : "v"(loff), "v"(gaddr)
               : "memory");
}
DEVINL void wait_async() { asm volatile("s_wait_asynccnt 0" ::: "memory"); }

// A fragment: 16x32 bf16, row-major source (row = lane&15).
// VGPR v holds K = 16*(lane>=16) + 2*v + {0,1}.
// Per lane this is two contiguous 16B chunks -> 2x ds_load_b128.
DEVINL v16bf load_frag_a(const __bf16* A, int lda, int lane) {
  int row = lane & 15;
  int hi  = (lane >> 4) & 1;
  const __bf16* p = A + row * lda;
  v16bf f;
#pragma unroll
  for (int v = 0; v < 8; ++v) {
    int k = hi * 16 + v * 2;
    f[2 * v]     = p[k];
    f[2 * v + 1] = p[k + 1];
  }
  return f;
}

// B fragment from pre-swizzled storage: base[(lane)*16 + k'] holds
// B[16*(lane>=16) + k'][lane&15] -> one contiguous 32B read per lane.
DEVINL v16bf load_frag_sw(const __bf16* base, int lane) {
  const __bf16* p = base + (lane << 4);
  v16bf f;
#pragma unroll
  for (int e = 0; e < 16; ++e) f[e] = p[e];
  return f;
}

// B fragment where the logical B = Kt^T (Kt row-major [n][k]); per-lane reads
// are contiguous along k -> 2x ds_load_b128.
DEVINL v16bf load_frag_bT(const __bf16* Kt, int ldk, int lane) {
  int n  = lane & 15;
  int hi = (lane >> 4) & 1;
  const __bf16* p = Kt + n * ldk + hi * 16;
  v16bf f;
#pragma unroll
  for (int v = 0; v < 8; ++v) {
    f[2 * v]     = p[2 * v];
    f[2 * v + 1] = p[2 * v + 1];
  }
  return f;
}

// ---------------------------------------------------------------------------
// Kernel 1: input projection  out = X[8192,K] @ W[K,512]  (K = 84 or 50)
// Writes fp32 (residual) and bf16 (GEMM operand) copies.
// ---------------------------------------------------------------------------
__global__ __launch_bounds__(256) void proj_in_kernel(const float* __restrict__ X,
                                                      const float* __restrict__ W,
                                                      float* __restrict__ outF,
                                                      __bf16* __restrict__ outB,
                                                      int K) {
  __shared__ float xs[96];
  int r = blockIdx.x;
  if ((int)threadIdx.x < K) xs[threadIdx.x] = X[(size_t)r * K + threadIdx.x];
  __syncthreads();
  for (int c = threadIdx.x; c < 512; c += 256) {
    float acc = 0.f;
    for (int k = 0; k < K; ++k) acc = fmaf(xs[k], W[(size_t)k * 512 + c], acc);
    outF[(size_t)r * 512 + c] = acc;
    outB[(size_t)r * 512 + c] = (__bf16)acc;
  }
}

// ---------------------------------------------------------------------------
// Generic WMMA GEMM:  C[M,N] = A_bf16[M,K] @ W_f32[K,N]   (epilogue templated)
// Block tile 128x64, BK=32, 8 waves (4x2), each wave 32x32 (2x2 frags).
// A staged via async DMA; B (fp32->bf16) staged in swizzled fragment order.
// ---------------------------------------------------------------------------
struct EpiQKV {  // scatter 1536 cols into q/k [B,H,N,64] and v TRANSPOSED [B,H,64,N]
  __bf16 *q, *k, *v;
  DEVINL void operator()(int r, int c, float val) const {
    int b = r >> 10, n = r & 1023;
    int s = c >> 9, h = (c >> 6) & 7, e = c & 63;
    if (s == 2) {  // v: [B,H,d,N] so attention V staging reads contiguous n
      v[((size_t)((b << 3) + h) * 64 + e) * 1024 + n] = (__bf16)val;
    } else {
      __bf16* dst = (s == 0) ? q : k;
      dst[((size_t)((b << 3) + h) * 1024 + n) * 64 + e] = (__bf16)val;
    }
  }
};

struct EpiQProj {  // [B*H*N,128] bf16; optional fp32 copy in [B,N,1024] layout
  __bf16* out;
  float*  f32out;
  DEVINL void operator()(int r, int c, float val) const {
    out[(size_t)r * 128 + c] = (__bf16)val;
    if (f32out) {
      int b = r >> 13, h = (r >> 10) & 7, n = r & 1023;
      f32out[((size_t)(b << 10) + n) * 1024 + h * 128 + c] = val;
    }
  }
};

struct EpiOut {  // out[r,colOff+c] = acc + bias + residual + q1r
  float*       out;
  const float* bias;
  const float* resid;
  const float* qadd;
  int          colOff;
  DEVINL void operator()(int r, int c, float val) const {
    float x = val + bias[c] + resid[(size_t)r * 512 + c];
    size_t oi = (size_t)r * 1024 + colOff + c;
    out[oi] = x + qadd[oi];
  }
};

template <class Epi>
__global__ __launch_bounds__(256) void gemm_bf16_kernel(const __bf16* __restrict__ A,
                                                        const float* __restrict__ W,
                                                        int M, int K, int N, Epi epi) {
  __shared__ __bf16 As[128][40];          // row-major + pad (A frags merge to b128)
  __shared__ __bf16 Bsw[4 * 32 * 16];     // swizzled: [ct][lane][k'] (4 KiB)
  int tid = threadIdx.x, lane = tid & 31, wave = tid >> 5;
  int hi = (lane >> 4) & 1, ln = lane & 15;
  int nTiles = N >> 6;
  int row0 = (blockIdx.x / nTiles) * 128;
  int col0 = (blockIdx.x % nTiles) * 64;
  int wm = wave >> 1, wn = wave & 1;

  v8f acc[2][2];
#pragma unroll
  for (int i = 0; i < 2; ++i)
#pragma unroll
    for (int j = 0; j < 2; ++j)
#pragma unroll
      for (int e = 0; e < 8; ++e) acc[i][j][e] = 0.f;

  for (int k0 = 0; k0 < K; k0 += 32) {
    // A tile: 128x32 bf16 via async DMA (2x b128 per thread)
    {
      int r = tid >> 1, c = (tid & 1) << 4;
      const __bf16* src = A + (size_t)(row0 + r) * K + k0 + c;
      async_load_b128(&As[r][c], src);
      async_load_b128(&As[r][c + 8], src + 8);
      if (k0 + 32 < K) __builtin_prefetch(src + 32, 0, 0);  // global_prefetch_b8
    }
    // B tile: 32x64 fp32 -> bf16, written in fragment order.
    // Task t: column n (0..63), k-quad kq (0..7). 4 strided f32 loads
    // (coalesced across n), one packed uint2 LDS store.
#pragma unroll
    for (int it = 0; it < 2; ++it) {
      int t = tid + it * 256;
      int n = t & 63, kq = t >> 6;
      int k = kq << 2;
      const float* wp = &W[(size_t)(k0 + k) * N + col0 + n];
      alignas(8) __bf16 t4[4];
#pragma unroll
      for (int i = 0; i < 4; ++i) t4[i] = (__bf16)wp[(size_t)i * N];
      int bhi = k >> 4, kp = k & 15;
      *(uint2*)&Bsw[((((n >> 4) * 32) + bhi * 16 + (n & 15)) << 4) + kp] =
          *(const uint2*)t4;
    }
    wait_async();
    __syncthreads();

    v16bf bfr[2];
#pragma unroll
    for (int j = 0; j < 2; ++j)
      bfr[j] = load_frag_sw(&Bsw[((wn * 2 + j) * 32) << 4], lane);
#pragma unroll
    for (int i = 0; i < 2; ++i) {
      v16bf a = load_frag_a(&As[wm * 32 + i * 16][0], 40, lane);
#pragma unroll
      for (int j = 0; j < 2; ++j) acc[i][j] = wmma_bf16(a, bfr[j], acc[i][j]);
    }
    __syncthreads();
  }

#pragma unroll
  for (int i = 0; i < 2; ++i)
#pragma unroll
    for (int j = 0; j < 2; ++j)
#pragma unroll
      for (int v = 0; v < 8; ++v) {
        int r = row0 + wm * 32 + i * 16 + hi * 8 + v;
        int c = col0 + wn * 32 + j * 16 + ln;
        epi(r, c, acc[i][j][v]);
      }
}

// ---------------------------------------------------------------------------
// Flash attention:  softmax(Q Kt^T * 0.125) @ V  per (b,h), 128 q-rows/block.
// Q: [B*H, 1024, 128] bf16.  K supplied as two 64-wide halves [B,H,N,*];
// V supplied TRANSPOSED [B,H,d,N].  K/V tiles staged with async DMA.
// VDIM = 128 (branch 1) or 64 (branch 2).
// ---------------------------------------------------------------------------
template <int VDIM>
__global__ __launch_bounds__(256) void attention_kernel(
    const __bf16* __restrict__ Q, const __bf16* __restrict__ K0, int ldK0,
    const __bf16* __restrict__ K1, int ldK1, const __bf16* __restrict__ V0,
    const __bf16* __restrict__ V1, __bf16* __restrict__ outp, int outStride) {
  constexpr int NQ = 1024;
  constexpr int LDK = 136, LDP = 72;  // 272B rows: 16B-aligned for b128
  constexpr int VTILES = VDIM / 16;
  __shared__ __bf16 Ks[64 * LDK];                  // row-major [n][128]
  __shared__ __bf16 Vsw[2 * VTILES * 32 * 16];     // swizzled frag order
  __shared__ __bf16 Ps[8 * 16 * LDP];

  int tid = threadIdx.x, lane = tid & 31, wave = tid >> 5;
  int hi = (lane >> 4) & 1, ln = lane & 15;
  int qt = blockIdx.x & 7;
  int bh = blockIdx.x >> 3;
  int b = bh >> 3, h = bh & 7;
  int qbase = qt * 128;

  const __bf16* k0p = K0 + (size_t)bh * NQ * ldK0;
  const __bf16* k1p = K1 + (size_t)bh * NQ * ldK1;
  const __bf16* v0p = V0 + (size_t)bh * 64 * NQ;  // [d][N]
  const __bf16* v1p = V1 + (size_t)bh * 64 * NQ;

  // Q fragments for this wave's 16 query rows (held in VGPRs for all 16 k-tiles)
  v16bf qf[4];
  {
    const __bf16* qrow = Q + ((size_t)bh * NQ + qbase + wave * 16 + ln) * 128;
#pragma unroll
    for (int kd = 0; kd < 4; ++kd)
#pragma unroll
      for (int v = 0; v < 8; ++v) {
        int k = kd * 32 + hi * 16 + v * 2;
        qf[kd][2 * v]     = qrow[k];
        qf[kd][2 * v + 1] = qrow[k + 1];
      }
  }

  v8f o[VTILES];
#pragma unroll
  for (int j = 0; j < VTILES; ++j)
#pragma unroll
    for (int e = 0; e < 8; ++e) o[j][e] = 0.f;
  float mrun[8], lrun[8];
#pragma unroll
  for (int v = 0; v < 8; ++v) { mrun[v] = -1e30f; lrun[v] = 0.f; }

  for (int kt = 0; kt < NQ / 64; ++kt) {
    int krow0 = kt * 64;
    // K tile [64 x 128]: 16B async chunks (halves are 8-elem aligned)
    for (int g = tid; g < 64 * 16; g += 256) {
      int r = g >> 4, c = (g & 15) << 3;
      const __bf16* src = (c < 64) ? (k0p + (size_t)(krow0 + r) * ldK0 + c)
                                   : (k1p + (size_t)(krow0 + r) * ldK1 + (c - 64));
      async_load_b128(&Ks[r * LDK + c], src);
    }
    // V tile staged swizzled via async DMA: task = (dim c, n-octet q);
    // transposed global V makes 8 consecutive k-rows one contiguous b128.
    for (int g = tid; g < VDIM * 8; g += 256) {
      int c = g >> 3, q = g & 7;
      int kr = q << 3;
      const __bf16* src = (c < 64) ? (v0p + (size_t)c * NQ + krow0 + kr)
                                   : (v1p + (size_t)(c - 64) * NQ + krow0 + kr);
      int kc = kr >> 5, k5 = kr & 31, vhi = k5 >> 4, kp = k5 & 15;
      int j = c >> 4, col = c & 15;
      async_load_b128(&Vsw[(((kc * VTILES + j) * 32 + vhi * 16 + col) << 4) + kp],
                      src);
    }
    wait_async();
    __syncthreads();

    // S = Q @ K^T   (16 q rows x 64 k cols per wave)
    v8f s[4];
#pragma unroll
    for (int ct = 0; ct < 4; ++ct)
#pragma unroll
      for (int e = 0; e < 8; ++e) s[ct][e] = 0.f;
#pragma unroll
    for (int kd = 0; kd < 4; ++kd)
#pragma unroll
      for (int ct = 0; ct < 4; ++ct) {
        v16bf bfrag = load_frag_bT(&Ks[(ct * 16) * LDK + kd * 32], LDK, lane);
        s[ct] = wmma_bf16(qf[kd], bfrag, s[ct]);
      }

    // Online softmax (rows 0-7 in lanes 0-15, rows 8-15 in lanes 16-31;
    // xor-shuffles 1/2/4/8 reduce each 16-lane half independently)
    float mnew[8], corr[8], ladd[8];
#pragma unroll
    for (int v = 0; v < 8; ++v) {
      float mx = -1e30f;
#pragma unroll
      for (int ct = 0; ct < 4; ++ct) {
        s[ct][v] *= 0.125f;
        mx = fmaxf(mx, s[ct][v]);
      }
      mx = fmaxf(mx, __shfl_xor(mx, 1, 32));
      mx = fmaxf(mx, __shfl_xor(mx, 2, 32));
      mx = fmaxf(mx, __shfl_xor(mx, 4, 32));
      mx = fmaxf(mx, __shfl_xor(mx, 8, 32));
      mnew[v] = fmaxf(mrun[v], mx);
      corr[v] = __expf(mrun[v] - mnew[v]);
      mrun[v] = mnew[v];
      ladd[v] = 0.f;
    }
#pragma unroll
    for (int ct = 0; ct < 4; ++ct)
#pragma unroll
      for (int v = 0; v < 8; ++v) {
        float p = __expf(s[ct][v] - mnew[v]);
        s[ct][v] = p;
        ladd[v] += p;
      }
#pragma unroll
    for (int v = 0; v < 8; ++v) {
      ladd[v] += __shfl_xor(ladd[v], 1, 32);
      ladd[v] += __shfl_xor(ladd[v], 2, 32);
      ladd[v] += __shfl_xor(ladd[v], 4, 32);
      ladd[v] += __shfl_xor(ladd[v], 8, 32);
      lrun[v] = lrun[v] * corr[v] + ladd[v];
    }
#pragma unroll
    for (int j = 0; j < VTILES; ++j)
#pragma unroll
      for (int v = 0; v < 8; ++v) o[j][v] *= corr[v];

    // Re-fragment P via per-wave LDS (C layout -> A layout)
    __bf16* Pw = &Ps[wave * 16 * LDP];
#pragma unroll
    for (int ct = 0; ct < 4; ++ct)
#pragma unroll
      for (int v = 0; v < 8; ++v)
        Pw[(hi * 8 + v) * LDP + ct * 16 + ln] = (__bf16)s[ct][v];
    asm volatile("s_wait_dscnt 0" ::: "memory");  // own-wave LDS RAW ordering

    // O += P @ V   (V fragments are contiguous ds_load_b128 pairs)
#pragma unroll
    for (int kc = 0; kc < 2; ++kc) {
      v16bf pa = load_frag_a(Pw + kc * 32, LDP, lane);
#pragma unroll
      for (int j = 0; j < VTILES; ++j) {
        v16bf vb = load_frag_sw(&Vsw[((kc * VTILES + j) * 32) << 4], lane);
        o[j] = wmma_bf16(pa, vb, o[j]);
      }
    }
    __syncthreads();
  }

#pragma unroll
  for (int v = 0; v < 8; ++v) lrun[v] = 1.f / lrun[v];
#pragma unroll
  for (int j = 0; j < VTILES; ++j)
#pragma unroll
    for (int v = 0; v < 8; ++v) {
      int n = qbase + wave * 16 + hi * 8 + v;
      int c = j * 16 + ln;
      outp[((size_t)b * NQ + n) * outStride + h * VDIM + c] =
          (__bf16)(o[j][v] * lrun[v]);
    }
}

// ---------------------------------------------------------------------------
// Host-side orchestration
// ---------------------------------------------------------------------------
extern "C" void kernel_launch(void* const* d_in, const int* in_sizes, int n_in,
                              void* d_out, int out_size, void* d_ws, size_t ws_size,
                              hipStream_t stream) {
  (void)in_sizes; (void)n_in; (void)out_size; (void)ws_size;
  const float* x     = (const float*)d_in[0];
  const float* y     = (const float*)d_in[1];
  const float* W1    = (const float*)d_in[2];
  const float* W2    = (const float*)d_in[3];
  const float* Wqkv1 = (const float*)d_in[4];
  const float* Wqkv2 = (const float*)d_in[5];
  const float* Wq1   = (const float*)d_in[6];
  const float* Wq2   = (const float*)d_in[7];
  const float* Wk2   = (const float*)d_in[8];
  const float* Wp1   = (const float*)d_in[9];
  const float* bp1   = (const float*)d_in[10];
  const float* Wp2   = (const float*)d_in[11];
  const float* bp2   = (const float*)d_in[12];
  float* out = (float*)d_out;
  char*  ws  = (char*)d_ws;

  constexpr size_t SZ_F32 = 8192ull * 512 * 4;   // 16 MiB
  constexpr size_t SZ_BF  = 8192ull * 512 * 2;   //  8 MiB
  constexpr size_t SZ_HD  = 65536ull * 64 * 2;   //  8 MiB  [B,H,N,64] (v: [B,H,64,N])
  constexpr size_t SZ_P   = 65536ull * 128 * 2;  // 16 MiB  [B,H,N,128] bf16
  constexpr size_t SZ_Q1R = 8192ull * 1024 * 4;  // 32 MiB
  constexpr size_t SZ_O1  = 8192ull * 1024 * 2;  // 16 MiB
  size_t off = 0;
  float*  Xf  = (float*)(ws + off);  off += SZ_F32;
  float*  Yf  = (float*)(ws + off);  off += SZ_F32;
  __bf16* Xb  = (__bf16*)(ws + off); off += SZ_BF;
  __bf16* Yb  = (__bf16*)(ws + off); off += SZ_BF;
  __bf16* q1b = (__bf16*)(ws + off); off += SZ_HD;
  __bf16* k1b = (__bf16*)(ws + off); off += SZ_HD;
  __bf16* v1t = (__bf16*)(ws + off); off += SZ_HD;
  __bf16* q2b = (__bf16*)(ws + off); off += SZ_HD;
  __bf16* k2b = (__bf16*)(ws + off); off += SZ_HD;
  __bf16* v2t = (__bf16*)(ws + off); off += SZ_HD;
  __bf16* q1p = (__bf16*)(ws + off); off += SZ_P;
  __bf16* q2p = (__bf16*)(ws + off); off += SZ_P;
  __bf16* k2p = (__bf16*)(ws + off); off += SZ_P;
  float*  q1r = (float*)(ws + off);  off += SZ_Q1R;
  __bf16* o1  = (__bf16*)(ws + off); off += SZ_O1;
  __bf16* o2  = (__bf16*)(ws + off);

  // 1) input projections (fp32 residual + bf16 operand)
  proj_in_kernel<<<8192, 256, 0, stream>>>(x, W1, Xf, Xb, 84);
  proj_in_kernel<<<8192, 256, 0, stream>>>(y, W2, Yf, Yb, 50);

  // 2) QKV GEMMs, scatter-split (q/k row-major, v transposed)
  gemm_bf16_kernel<EpiQKV><<<64 * 24, 256, 0, stream>>>(Xb, Wqkv1, 8192, 512, 1536,
                                                        EpiQKV{q1b, k1b, v1t});
  gemm_bf16_kernel<EpiQKV><<<64 * 24, 256, 0, stream>>>(Yb, Wqkv2, 8192, 512, 1536,
                                                        EpiQKV{q2b, k2b, v2t});

  // 3) q/k projections to 2d=128 (q1 also materialized fp32 for final add)
  gemm_bf16_kernel<EpiQProj><<<512 * 2, 256, 0, stream>>>(q1b, Wq1, 65536, 64, 128,
                                                          EpiQProj{q1p, q1r});
  gemm_bf16_kernel<EpiQProj><<<512 * 2, 256, 0, stream>>>(q2b, Wq2, 65536, 64, 128,
                                                          EpiQProj{q2p, nullptr});
  gemm_bf16_kernel<EpiQProj><<<512 * 2, 256, 0, stream>>>(k2b, Wk2, 65536, 64, 128,
                                                          EpiQProj{k2p, nullptr});

  // 4) attention (branch 1: K/V = concat halves; branch 2: K = k2p, V = v2)
  attention_kernel<128><<<512, 256, 0, stream>>>(q1p, k1b, 64, k2b, 64, v1t, v2t,
                                                 o1, 1024);
  attention_kernel<64><<<512, 256, 0, stream>>>(q2p, k2p, 128, k2p + 64, 128, v2t,
                                                v2t, o2, 512);

  // 5) output projections + residual + bias + q1r (cover cols 0-511 / 512-1023)
  gemm_bf16_kernel<EpiOut><<<64 * 8, 256, 0, stream>>>(o1, Wp1, 8192, 1024, 512,
                                                       EpiOut{out, bp1, Xf, q1r, 0});
  gemm_bf16_kernel<EpiOut><<<64 * 8, 256, 0, stream>>>(o2, Wp2, 8192, 512, 512,
                                                       EpiOut{out, bp2, Yf, q1r, 512});
}
